// AttentionDecoder_55490977464701
// MI455X (gfx1250) — matrix operand order repaired and checked by reading, not compiled, not run on hardware
//
#include <hip/hip_runtime.h>
#include <math.h>

#define Bdim 64
#define Sdim 512
#define Hdim 1024
#define Vdim 32000

typedef __attribute__((ext_vector_type(16))) __bf16 v16bf;
typedef __attribute__((ext_vector_type(8)))  __bf16 bf16x8;
typedef __attribute__((ext_vector_type(8)))  float  v8f;

union Pk4 { __bf16 h[4]; unsigned long long u; };

__device__ __forceinline__ v16bf cat8(bf16x8 lo, bf16x8 hi) {
    return __builtin_shufflevector(lo, hi, 0,1,2,3,4,5,6,7,8,9,10,11,12,13,14,15);
}

// ---------------------------------------------------------------------------
// 1) Convert attn_W1[:, :H] (fp32, row stride 2H) -> bf16 copy [H x H] row-major
// ---------------------------------------------------------------------------
__global__ __launch_bounds__(256)
void k_cvt_w1a(const float* __restrict__ W1, __bf16* __restrict__ w1a) {
    const int i = (blockIdx.x * 256 + threadIdx.x) * 4;   // element in H*H
    const int k = i >> 10, h = i & 1023;
    const float4 f = *(const float4*)(W1 + (size_t)k * (2 * Hdim) + h);
    Pk4 p;
    p.h[0] = (__bf16)f.x; p.h[1] = (__bf16)f.y;
    p.h[2] = (__bf16)f.z; p.h[3] = (__bf16)f.w;
    *(unsigned long long*)(w1a + i) = p.u;
}

// ---------------------------------------------------------------------------
// 2) Generic fp32 -> bf16 vector convert (hprev, xc activations)
// ---------------------------------------------------------------------------
__global__ __launch_bounds__(256)
void k_cvt_bf(const float* __restrict__ src, __bf16* __restrict__ dst) {
    const int i = (blockIdx.x * 256 + threadIdx.x) * 4;
    const float4 f = *(const float4*)(src + i);
    Pk4 p;
    p.h[0] = (__bf16)f.x; p.h[1] = (__bf16)f.y;
    p.h[2] = (__bf16)f.z; p.h[3] = (__bf16)f.w;
    *(unsigned long long*)(dst + i) = p.u;
}

// ---------------------------------------------------------------------------
// 3) Embedding gather -> xc[:, H:2H]
// ---------------------------------------------------------------------------
__global__ __launch_bounds__(256)
void k_embed(const int* __restrict__ x, const float* __restrict__ emb,
             float* __restrict__ xc) {
    const int idx = blockIdx.x * 256 + threadIdx.x;   // b*H + h
    const int b = idx >> 10, h = idx & 1023;
    xc[(size_t)b * (2 * Hdim) + Hdim + h] = emb[(size_t)x[b] * Hdim + h];
}

// ---------------------------------------------------------------------------
// 4) Generic skinny WMMA GEMM:  out[m][n] = sum_k A_bf16[m][k] * W[n][k] (+bias)
//    M = 64 (4 waves x one 16-row M-tile), 16 n-cols per block.
//    W fp32 rows are read ONCE, converted to bf16 into a double-buffered LDS
//    tile; next A-frag and next W-tile are prefetched one iteration ahead.
// ---------------------------------------------------------------------------
__global__ __launch_bounds__(128)
void k_gemm(const __bf16* __restrict__ A, int lda,
            const float* __restrict__ W, int ldw,
            const float* __restrict__ bias,
            float* __restrict__ out, int ldo, int K) {
    __shared__ __align__(16) __bf16 Bt[2][16 * 40];   // padded stride
    const int n0  = blockIdx.x * 16;
    const int tid = threadIdx.x;
    const int wave = tid >> 5, lane = tid & 31;
    const int hs = lane >> 4, l16 = lane & 15;
    const int nk = K >> 5;

    const int vrow = tid >> 3, hc = (tid & 7) * 4;
    const float*  wsrc = W + (size_t)(n0 + vrow) * ldw + hc;
    const __bf16* asrc = A + (size_t)(wave * 16 + l16) * lda + (hs ? 8 : 0);

    auto loadW4 = [&](int i) -> float4 {
        return *(const float4*)(wsrc + (size_t)i * 32);
    };
    auto loadA = [&](int i) -> v16bf {
        const __bf16* p = asrc + (size_t)i * 32;
        return cat8(*(const bf16x8*)p, *(const bf16x8*)(p + 16));
    };
    auto storeW = [&](int buf, float4 f) {
        Pk4 p;
        p.h[0] = (__bf16)f.x; p.h[1] = (__bf16)f.y;
        p.h[2] = (__bf16)f.z; p.h[3] = (__bf16)f.w;
        *(unsigned long long*)(&Bt[buf][vrow * 40 + hc]) = p.u;
    };

    v8f c = {};
    float4 wreg = loadW4(0);
    v16bf  acur = loadA(0);
    storeW(0, wreg);
    wreg = loadW4(nk > 1 ? 1 : 0);
    __syncthreads();

    for (int i = 0; i < nk; ++i) {
        const __bf16* bp = &Bt[i & 1][l16 * 40 + hs * 8];
        const v16bf bv = cat8(*(const bf16x8*)bp, *(const bf16x8*)(bp + 16));
        const v16bf anext = loadA(i + 1 < nk ? i + 1 : i);  // prefetch
        c = __builtin_amdgcn_wmma_f32_16x16x32_bf16(
                false, acur, false, bv, (short)0, c, false, false);
        __syncthreads();                       // all reads of Bt[i&1] done
        storeW((i + 1) & 1, wreg);             // wreg loaded 1 iter ago
        wreg = loadW4(i + 2 < nk ? i + 2 : 0); // prefetch next W tile
        acur = anext;
        __syncthreads();                       // stores visible
    }

    const int n = n0 + l16;
    const float bo = bias ? bias[n] : 0.f;
    #pragma unroll
    for (int r = 0; r < 8; ++r) {
        const int row = wave * 16 + (hs ? (r + 8) : r);
        out[(size_t)row * ldo + n] = c[r] + bo;
    }
}

// ---------------------------------------------------------------------------
// 5) Fused attention-score kernel (the 68.7-GFLOP GEMM).
//    One block per 16 (b,s) rows; A slab bf16 in LDS; B-frags from the
//    L2-resident bf16 W1 copy with one-iteration register rotation so the
//    WMMA never waits on a just-issued global load.
// ---------------------------------------------------------------------------
__global__ __launch_bounds__(256)
void k_attn_scores(const float* __restrict__ ann,    // B*S*H fp32
                   const __bf16* __restrict__ w1a,   // H x H bf16
                   const float* __restrict__ hB,     // B*H
                   const float* __restrict__ b1,     // H
                   const float* __restrict__ w2,     // H
                   float* __restrict__ scores)       // B*S (pre-zeroed)
{
    __shared__ __align__(16) __bf16 Alds[16 * 1048];  // padded stride
    const int bs0 = blockIdx.x * 16;
    const int tid = threadIdx.x;

    // Stage A slab: 16 rows x 1024 cols fp32 -> bf16 LDS
    for (int i = tid * 4; i < 16 * 1024; i += 256 * 4) {
        const int m = i >> 10, h = i & 1023;
        const float4 f = *(const float4*)(ann + (size_t)(bs0 + m) * Hdim + h);
        Pk4 p;
        p.h[0] = (__bf16)f.x; p.h[1] = (__bf16)f.y;
        p.h[2] = (__bf16)f.z; p.h[3] = (__bf16)f.w;
        *(unsigned long long*)(&Alds[m * 1048 + h]) = p.u;
    }
    __syncthreads();

    const int wave = tid >> 5;
    const int lane = tid & 31;
    const int hs   = lane >> 4;
    const int l16  = lane & 15;
    const int b    = bs0 / Sdim;   // 16-row tiles never cross a b boundary

    float sacc[8];
    #pragma unroll
    for (int r = 0; r < 8; ++r) sacc[r] = 0.f;

    for (int kt = 0; kt < 8; ++kt) {
        const int k0 = (wave * 8 + kt) * 16;
        const __bf16* bbase = w1a + (size_t)(k0 + l16) * Hdim + hs * 8;
        v8f c = {};
        bf16x8 b0 = *(const bf16x8*)(bbase);
        bf16x8 b1v = *(const bf16x8*)(bbase + 16);
        #pragma unroll 4
        for (int h0 = 0; h0 < Hdim; h0 += 32) {
            const int hn = (h0 + 32 < Hdim) ? (h0 + 32) : 0;   // clamped prefetch
            const bf16x8 nb0 = *(const bf16x8*)(bbase + hn);
            const bf16x8 nb1 = *(const bf16x8*)(bbase + hn + 16);
            const __bf16* ap = &Alds[l16 * 1048 + h0 + (hs ? 8 : 0)];
            const v16bf a = cat8(*(const bf16x8*)ap, *(const bf16x8*)(ap + 16));
            c = __builtin_amdgcn_wmma_f32_16x16x32_bf16(
                    false, a, false, cat8(b0, b1v), (short)0, c, false, false);
            b0 = nb0; b1v = nb1;
        }
        const int kcol = k0 + l16;
        const float addc = hB[(size_t)b * Hdim + kcol] + b1[kcol];
        const float w2v  = w2[kcol];
        #pragma unroll
        for (int r = 0; r < 8; ++r) {
            float v = fmaxf(c[r] + addc, 0.f) * w2v;
            v += __shfl_xor(v, 1, 32);
            v += __shfl_xor(v, 2, 32);
            v += __shfl_xor(v, 4, 32);
            v += __shfl_xor(v, 8, 32);
            sacc[r] += v;
        }
    }
    if (l16 == 0) {
        #pragma unroll
        for (int r = 0; r < 8; ++r) {
            const int row = hs ? (r + 8) : r;
            atomicAdd(&scores[bs0 + row], sacc[r]);
        }
    }
}

// ---------------------------------------------------------------------------
// 6) Softmax over S per batch row. (b2 is a constant shift -> softmax-invariant.)
// ---------------------------------------------------------------------------
__global__ __launch_bounds__(256)
void k_softmax(const float* __restrict__ scores, float* __restrict__ attnw,
               float* __restrict__ out_attn) {
    const int b = blockIdx.x, tid = threadIdx.x;
    __shared__ float red[256];
    const float v0 = scores[(size_t)b * Sdim + tid];
    const float v1 = scores[(size_t)b * Sdim + 256 + tid];
    red[tid] = fmaxf(v0, v1);
    __syncthreads();
    for (int off = 128; off > 0; off >>= 1) {
        if (tid < off) red[tid] = fmaxf(red[tid], red[tid + off]);
        __syncthreads();
    }
    const float m = red[0];
    __syncthreads();
    const float e0 = __expf(v0 - m), e1 = __expf(v1 - m);
    red[tid] = e0 + e1;
    __syncthreads();
    for (int off = 128; off > 0; off >>= 1) {
        if (tid < off) red[tid] += red[tid + off];
        __syncthreads();
    }
    const float inv = 1.0f / red[0];
    const float w0 = e0 * inv, w1 = e1 * inv;
    attnw[(size_t)b * Sdim + tid]        = w0;
    attnw[(size_t)b * Sdim + 256 + tid]  = w1;
    out_attn[(size_t)b * Sdim + tid]       = w0;
    out_attn[(size_t)b * Sdim + 256 + tid] = w1;
}

// ---------------------------------------------------------------------------
// 7) context[b][h] = sum_s w[b][s] * annotations[b][s][h]  -> xc[:, :H]
// ---------------------------------------------------------------------------
__global__ __launch_bounds__(256)
void k_context(const float* __restrict__ ann, const float* __restrict__ attnw,
               float* __restrict__ xc) {
    const int b  = blockIdx.x >> 2;
    const int h  = (blockIdx.x & 3) * 256 + threadIdx.x;
    __shared__ float wl[Sdim];
    for (int s = threadIdx.x; s < Sdim; s += 256)
        wl[s] = attnw[(size_t)b * Sdim + s];
    __syncthreads();
    const float* __restrict__ ap = ann + ((size_t)b * Sdim) * Hdim + h;
    float acc = 0.f;
    #pragma unroll 4
    for (int s = 0; s < Sdim; ++s) {
        if ((s & 7) == 0 && s + 16 < Sdim)
            __builtin_prefetch(ap + (size_t)(s + 16) * Hdim, 0, 0);
        acc = fmaf(wl[s], ap[(size_t)s * Hdim], acc);
    }
    xc[(size_t)b * (2 * Hdim) + h] = acc;
}

// ---------------------------------------------------------------------------
// 8) Elementwise GRU gates from the 5 GEMM partials -> h_new (fp32 + bf16)
// ---------------------------------------------------------------------------
__global__ __launch_bounds__(256)
void k_gates(const float* __restrict__ giz, const float* __restrict__ gir,
             const float* __restrict__ gin, const float* __restrict__ ghz,
             const float* __restrict__ ghn,
             const float* __restrict__ b_iz, const float* __restrict__ b_ir,
             const float* __restrict__ b_in, const float* __restrict__ b_hz,
             const float* __restrict__ b_hn,
             const float* __restrict__ hprev,
             float* __restrict__ hnew_out, __bf16* __restrict__ hnew_bf) {
    const int idx = blockIdx.x * 256 + threadIdx.x;   // b*H + k
    const int k = idx & 1023;
    const float hzb = ghz[idx] + b_hz[k];
    const float z = 1.f / (1.f + __expf(-(giz[idx] + b_iz[k] + hzb)));
    const float r = 1.f / (1.f + __expf(-(gir[idx] + b_ir[k] + hzb)));
    const float g = tanhf(gin[idx] + b_in[k] + r * (ghn[idx] + b_hn[k]));
    const float hnv = (1.f - z) * g + z * hprev[idx];
    hnew_out[idx] = hnv;
    hnew_bf[idx]  = (__bf16)hnv;
}

// ---------------------------------------------------------------------------
// Launch
// ---------------------------------------------------------------------------
extern "C" void kernel_launch(void* const* d_in, const int* in_sizes, int n_in,
                              void* d_out, int out_size, void* d_ws, size_t ws_size,
                              hipStream_t stream) {
    const int*   x        = (const int*)  d_in[0];
    const float* h_prev   = (const float*)d_in[1];
    const float* ann      = (const float*)d_in[2];
    const float* emb      = (const float*)d_in[3];
    const float* attn_W1  = (const float*)d_in[4];
    const float* attn_b1  = (const float*)d_in[5];
    const float* attn_W2  = (const float*)d_in[6];
    const float* W_ir     = (const float*)d_in[8];
    const float* b_ir     = (const float*)d_in[9];
    const float* W_iz     = (const float*)d_in[10];
    const float* b_iz     = (const float*)d_in[11];
    const float* W_in     = (const float*)d_in[12];
    const float* b_in     = (const float*)d_in[13];
    const float* W_hz     = (const float*)d_in[14];
    const float* b_hz     = (const float*)d_in[15];
    const float* W_hn     = (const float*)d_in[16];
    const float* b_hn     = (const float*)d_in[17];
    const float* W_out    = (const float*)d_in[18];
    const float* b_out    = (const float*)d_in[19];

    // d_out layout: output (B*V) | h_new (B*H) | attn_weights (B*S)
    float* out_logits = (float*)d_out;
    float* out_hnew   = out_logits + (size_t)Bdim * Vdim;
    float* out_attn   = out_hnew + (size_t)Bdim * Hdim;

    // ws layout
    char* ws = (char*)d_ws;
    float*  scores   = (float*)(ws);                   // B*S*4       = 131072
    float*  attnw    = (float*)(ws + 131072);          // B*S*4
    float*  hB       = (float*)(ws + 262144);          // B*H*4       = 262144
    float*  xc       = (float*)(ws + 524288);          // B*2H*4      = 524288
    __bf16* w1a      = (__bf16*)(ws + 1048576);        // H*H*2       = 2097152
    __bf16* hnew_bf  = (__bf16*)(ws + 3145728);        // B*H*2       = 131072
    __bf16* hprev_bf = (__bf16*)(ws + 3276800);        // B*H*2       = 131072
    __bf16* xc_bf    = (__bf16*)(ws + 3407872);        // B*2H*2      = 262144
    float*  giz      = (float*)(ws + 3670016);         // B*H*4
    float*  gir      = (float*)(ws + 3932160);         // B*H*4
    float*  gin      = (float*)(ws + 4194304);         // B*H*4
    float*  ghz      = (float*)(ws + 4456448);         // B*H*4
    float*  ghn      = (float*)(ws + 4718592);         // B*H*4  -> total ~4.75MB

    hipMemsetAsync(scores, 0, (size_t)Bdim * Sdim * sizeof(float), stream);

    k_cvt_w1a<<<(Hdim * Hdim) / 1024, 256, 0, stream>>>(attn_W1, w1a);
    k_cvt_bf <<<(Bdim * Hdim) / 1024, 256, 0, stream>>>(h_prev, hprev_bf);

    // hB = h_prev @ attn_W1[:, H:].T   (K=1024, W rows stride 2H, offset H)
    k_gemm<<<Hdim / 16, 128, 0, stream>>>(hprev_bf, Hdim, attn_W1 + Hdim,
                                          2 * Hdim, nullptr, hB, Hdim, Hdim);

    k_embed<<<(Bdim * Hdim) / 256, 256, 0, stream>>>(x, emb, xc);

    k_attn_scores<<<(Bdim * Sdim) / 16, 256, 0, stream>>>(ann, w1a, hB, attn_b1,
                                                          attn_W2, scores);
    k_softmax<<<Bdim, 256, 0, stream>>>(scores, attnw, out_attn);
    k_context<<<Bdim * 4, 256, 0, stream>>>(ann, attnw, xc);

    k_cvt_bf<<<(Bdim * 2 * Hdim) / 1024, 256, 0, stream>>>(xc, xc_bf);

    // GRU input/hidden projections as single-pass WMMA GEMMs
    k_gemm<<<Hdim / 16, 128, 0, stream>>>(xc_bf, 2 * Hdim, W_iz, 2 * Hdim,
                                          nullptr, giz, Hdim, 2 * Hdim);
    k_gemm<<<Hdim / 16, 128, 0, stream>>>(xc_bf, 2 * Hdim, W_ir, 2 * Hdim,
                                          nullptr, gir, Hdim, 2 * Hdim);
    k_gemm<<<Hdim / 16, 128, 0, stream>>>(xc_bf, 2 * Hdim, W_in, 2 * Hdim,
                                          nullptr, gin, Hdim, 2 * Hdim);
    k_gemm<<<Hdim / 16, 128, 0, stream>>>(hprev_bf, Hdim, W_hz, Hdim,
                                          nullptr, ghz, Hdim, Hdim);
    k_gemm<<<Hdim / 16, 128, 0, stream>>>(hprev_bf, Hdim, W_hn, Hdim,
                                          nullptr, ghn, Hdim, Hdim);

    k_gates<<<(Bdim * Hdim) / 256, 256, 0, stream>>>(giz, gir, gin, ghz, ghn,
                                                     b_iz, b_ir, b_in, b_hz, b_hn,
                                                     h_prev, out_hnew, hnew_bf);

    // logits = h_new @ W_out.T + b_out  (W_out fp32 read exactly once)
    k_gemm<<<Vdim / 16, 128, 0, stream>>>(hnew_bf, Hdim, W_out, Hdim,
                                          b_out, out_logits, Vdim, Hdim);
}
